// HoGN_23811298689149
// MI455X (gfx1250) — matrix operand order repaired and checked by reading, not compiled, not run on hardware
//
#include <hip/hip_runtime.h>
#include <hip/hip_bf16.h>

#define N_RNA  50000
#define N_NODE 100000   // N_RNA + N_PROT
#define FEAT   128
#define NEDGE  800000

typedef __attribute__((ext_vector_type(16))) __bf16 v16bf;
typedef __attribute__((ext_vector_type(8)))  __bf16 v8bf;
typedef __attribute__((ext_vector_type(4)))  __bf16 v4bf;
typedef __attribute__((ext_vector_type(8)))  float  v8f;
typedef __attribute__((ext_vector_type(4)))  float  v4f;

__device__ __forceinline__ v16bf bf_cat(v8bf a, v8bf b) {
  v16bf r;
#pragma unroll
  for (int i = 0; i < 8; ++i) { r[i] = a[i]; r[i + 8] = b[i]; }
  return r;
}

__device__ __forceinline__ v8f zero8() {
  v8f r;
#pragma unroll
  for (int i = 0; i < 8; ++i) r[i] = 0.0f;
  return r;
}

// B-fragment load: 2x8 bf16 at row (o0+n), k = c*32 + hf*8 and c*32 + 16 + hf*8
__device__ __forceinline__ v16bf load_bfrag(const __bf16* __restrict__ w,
                                            int row, int ldk, int c, int hf) {
  const __bf16* p = w + (long)row * ldk + c * 32 + hf * 8;
  return bf_cat(*(const v8bf*)p, *(const v8bf*)(p + 16));
}

// ---------------------------------------------------------------- weights->bf16
__global__ void k_cvt_weights(const float* Wl, const float* Wr,
                              const float* W1, const float* W2,
                              __bf16* wl, __bf16* wr, __bf16* w1, __bf16* w2) {
  int i = blockIdx.x * blockDim.x + threadIdx.x;
  if (i < 128 * 128) { wl[i] = (__bf16)Wl[i]; wr[i] = (__bf16)Wr[i]; }
  if (i < 64 * 128)  { w1[i] = (__bf16)W1[i]; }
  if (i < 32 * 64)   { w2[i] = (__bf16)W2[i]; }
}

// ---------------------------------------------------------------- zero agg+cnt
__global__ void k_zero(float* p, long n) {
  long i = (long)blockIdx.x * blockDim.x + threadIdx.x;
  if (i < n) p[i] = 0.0f;
}

// ------------------------------------------------- x[e] = n_fea[src]*n_fea[dst]
// one wave (32 lanes) per edge; lane handles 4 consecutive features
__global__ __launch_bounds__(256) void k_edge_feat(const float* __restrict__ rna,
                                                   const float* __restrict__ prot,
                                                   const int* __restrict__ ei,
                                                   __bf16* __restrict__ xbf) {
  long gt = (long)blockIdx.x * blockDim.x + threadIdx.x;
  long e = gt >> 5;
  int lane = threadIdx.x & 31;
  if (e >= NEDGE) return;
  int s = ei[e];
  int d = ei[NEDGE + e];
  const float* ps = (s < N_RNA) ? (rna  + (long)s * FEAT)
                                : (prot + (long)(s - N_RNA) * FEAT);
  const float* pd = (d < N_RNA) ? (rna  + (long)d * FEAT)
                                : (prot + (long)(d - N_RNA) * FEAT);
  int c = lane * 4;
  v4f a = *(const v4f*)(ps + c);
  v4f b = *(const v4f*)(pd + c);
  v4bf o;
#pragma unroll
  for (int i = 0; i < 4; ++i) o[i] = (__bf16)(a[i] * b[i]);
  *(v4bf*)(xbf + e * FEAT + c) = o;
}

// --------------------------------- agg[dst[e]] += x[src[e]] ; cnt[dst[e]] += 1
__global__ __launch_bounds__(256) void k_scatter(const __bf16* __restrict__ xbf,
                                                 const int* __restrict__ ei,
                                                 float* __restrict__ agg,
                                                 float* __restrict__ cnt) {
  long gt = (long)blockIdx.x * blockDim.x + threadIdx.x;
  long e = gt >> 5;
  int lane = threadIdx.x & 31;
  if (e >= NEDGE) return;
  long e2 = (long)ei[e];          // gather row of x (always < N_NODE)
  long d  = (long)ei[NEDGE + e];  // scatter destination (< N_NODE)
  int c = lane * 4;
  v4bf m = *(const v4bf*)(xbf + e2 * FEAT + c);
#pragma unroll
  for (int i = 0; i < 4; ++i)
    atomicAdd(&agg[d * FEAT + c + i], (float)m[i]);
  if (lane == 0) atomicAdd(&cnt[d], 1.0f);
}

// ------------------------------------------------------------- fused WMMA MLP
// One wave per 16-edge tile. 8 waves / block. bf16 WMMA, f32 accumulate.
__global__ __launch_bounds__(256) void k_fused(
    const __bf16* __restrict__ xbf, const float* __restrict__ agg,
    const float* __restrict__ cnt,
    const __bf16* __restrict__ wl, const __bf16* __restrict__ wr,
    const __bf16* __restrict__ w1, const __bf16* __restrict__ w2,
    const float* __restrict__ bl, const float* __restrict__ b1,
    const float* __restrict__ b2, const float* __restrict__ W3,
    const float* __restrict__ b3, float* __restrict__ out) {
  __shared__ alignas(16) __bf16 sh_h [8][16 * 128];
  __shared__ alignas(16) __bf16 sh_h1[8][16 * 64];
  __shared__ alignas(16) __bf16 sh_h2[8][16 * 32];

  int wave = threadIdx.x >> 5;
  int lane = threadIdx.x & 31;
  long tile = (long)blockIdx.x * 8 + wave;
  long e0 = tile * 16;
  if (e0 >= NEDGE) return;
  int mrow = lane & 15;   // A-matrix row / B-matrix column for this lane
  int hf   = lane >> 4;   // lane half selects K sub-block per ISA layout
  long row = e0 + mrow;

  // wave-uniform flag as SGPR -> scalar branch, EXEC stays all-1s for WMMA
  bool has_mean =
      __builtin_amdgcn_readfirstlane((int)(e0 < N_NODE)) != 0;

  // ---- A fragments of x (16x128 bf16, 4 K-chunks of 32) ----
  v16bf xa[4];
#pragma unroll
  for (int c = 0; c < 4; ++c) {
    const __bf16* p = xbf + row * FEAT + c * 32 + hf * 8;
    xa[c] = bf_cat(*(const v8bf*)p, *(const v8bf*)(p + 16));
  }

  // ---- A fragments of mean (zero for rows >= N_NODE; tile-uniform) ----
  v16bf ma[4];
  if (has_mean) {
    float s = 1.0f / fmaxf(cnt[row], 1.0f);
#pragma unroll
    for (int c = 0; c < 4; ++c) {
      const float* pa = agg + row * FEAT + c * 32 + hf * 8;
      v4f a0 = *(const v4f*)(pa);
      v4f a1 = *(const v4f*)(pa + 4);
      v4f a2 = *(const v4f*)(pa + 16);
      v4f a3 = *(const v4f*)(pa + 20);
#pragma unroll
      for (int u = 0; u < 4; ++u) {
        ma[c][u]      = (__bf16)(a0[u] * s);
        ma[c][4 + u]  = (__bf16)(a1[u] * s);
        ma[c][8 + u]  = (__bf16)(a2[u] * s);
        ma[c][12 + u] = (__bf16)(a3[u] * s);
      }
    }
  }

  // ---- layer 1: h = relu(mean@Wl^T + x@Wr^T + bl), [16x128] ----
#pragma unroll
  for (int nt = 0; nt < 8; ++nt) {
    // batch all B-fragment loads first so WMMAs overlap outstanding loads
    v16bf bfr[4], bfl[4];
#pragma unroll
    for (int c = 0; c < 4; ++c)
      bfr[c] = load_bfrag(wr, nt * 16 + mrow, 128, c, hf);
    if (has_mean) {
#pragma unroll
      for (int c = 0; c < 4; ++c)
        bfl[c] = load_bfrag(wl, nt * 16 + mrow, 128, c, hf);
    }
    v8f acc = zero8();
#pragma unroll
    for (int c = 0; c < 4; ++c)
      acc = __builtin_amdgcn_wmma_f32_16x16x32_bf16(false, xa[c], false, bfr[c],
                                                    (short)0, acc, false, false);
    if (has_mean) {
#pragma unroll
      for (int c = 0; c < 4; ++c)
        acc = __builtin_amdgcn_wmma_f32_16x16x32_bf16(false, ma[c], false, bfl[c],
                                                      (short)0, acc, false, false);
    }
    float bias = bl[nt * 16 + mrow];
#pragma unroll
    for (int r = 0; r < 8; ++r)
      sh_h[wave][(hf * 8 + r) * 128 + nt * 16 + mrow] =
          (__bf16)fmaxf(acc[r] + bias, 0.0f);
  }
  __syncthreads();

  // ---- layer 2: h1 = relu(h@W1^T + b1), [16x64] ----
  v16bf ha[4];
#pragma unroll
  for (int c = 0; c < 4; ++c) {
    const __bf16* p = &sh_h[wave][mrow * 128 + c * 32 + hf * 8];
    ha[c] = bf_cat(*(const v8bf*)p, *(const v8bf*)(p + 16));
  }
#pragma unroll
  for (int nt = 0; nt < 4; ++nt) {
    v16bf bw[4];
#pragma unroll
    for (int c = 0; c < 4; ++c)
      bw[c] = load_bfrag(w1, nt * 16 + mrow, 128, c, hf);
    v8f acc = zero8();
#pragma unroll
    for (int c = 0; c < 4; ++c)
      acc = __builtin_amdgcn_wmma_f32_16x16x32_bf16(false, ha[c], false, bw[c],
                                                    (short)0, acc, false, false);
    float bias = b1[nt * 16 + mrow];
#pragma unroll
    for (int r = 0; r < 8; ++r)
      sh_h1[wave][(hf * 8 + r) * 64 + nt * 16 + mrow] =
          (__bf16)fmaxf(acc[r] + bias, 0.0f);
  }
  __syncthreads();

  // ---- layer 3: h2 = relu(h1@W2^T + b2), [16x32] ----
  v16bf h1a[2];
#pragma unroll
  for (int c = 0; c < 2; ++c) {
    const __bf16* p = &sh_h1[wave][mrow * 64 + c * 32 + hf * 8];
    h1a[c] = bf_cat(*(const v8bf*)p, *(const v8bf*)(p + 16));
  }
#pragma unroll
  for (int nt = 0; nt < 2; ++nt) {
    v16bf bw[2];
#pragma unroll
    for (int c = 0; c < 2; ++c)
      bw[c] = load_bfrag(w2, nt * 16 + mrow, 64, c, hf);
    v8f acc = zero8();
#pragma unroll
    for (int c = 0; c < 2; ++c)
      acc = __builtin_amdgcn_wmma_f32_16x16x32_bf16(false, h1a[c], false, bw[c],
                                                    (short)0, acc, false, false);
    float bias = b2[nt * 16 + mrow];
#pragma unroll
    for (int r = 0; r < 8; ++r)
      sh_h2[wave][(hf * 8 + r) * 32 + nt * 16 + mrow] =
          (__bf16)fmaxf(acc[r] + bias, 0.0f);
  }
  __syncthreads();

  // ---- layer 4 + log_softmax (tiny, VALU): lanes 0..15, one row each ----
  if (lane < 16) {
    float l0 = b3[0], l1 = b3[1];
#pragma unroll
    for (int j = 0; j < 32; ++j) {
      float hv = (float)sh_h2[wave][lane * 32 + j];
      l0 += hv * W3[j];
      l1 += hv * W3[32 + j];
    }
    float mx  = fmaxf(l0, l1);
    float lse = mx + logf(expf(l0 - mx) + expf(l1 - mx));
    long e = e0 + lane;
    out[e * 2 + 0] = l0 - lse;
    out[e * 2 + 1] = l1 - lse;
  }
}

// ---------------------------------------------------------------- labels
__global__ void k_labels(float* __restrict__ out) {
  long e = (long)blockIdx.x * blockDim.x + threadIdx.x;
  if (e < NEDGE) out[2L * NEDGE + e] = (e < NEDGE / 2) ? 1.0f : 0.0f;
}

// ---------------------------------------------------------------- launcher
extern "C" void kernel_launch(void* const* d_in, const int* in_sizes, int n_in,
                              void* d_out, int out_size, void* d_ws, size_t ws_size,
                              hipStream_t stream) {
  const float* rna  = (const float*)d_in[0];
  const float* prot = (const float*)d_in[1];
  const int*   ei   = (const int*)d_in[2];
  const float* W_l  = (const float*)d_in[3];
  const float* b_l  = (const float*)d_in[4];
  const float* W_r  = (const float*)d_in[5];
  const float* W1   = (const float*)d_in[6];
  const float* b1   = (const float*)d_in[7];
  const float* W2   = (const float*)d_in[8];
  const float* b2   = (const float*)d_in[9];
  const float* W3   = (const float*)d_in[10];
  const float* b3   = (const float*)d_in[11];
  float* out = (float*)d_out;

  // workspace carve (all offsets 16B-aligned)
  char* ws = (char*)d_ws;
  size_t off = 0;
  __bf16* xbf = (__bf16*)(ws + off); off += (size_t)NEDGE * FEAT * sizeof(__bf16);
  float*  agg = (float*)(ws + off);  off += (size_t)N_NODE * FEAT * sizeof(float);
  float*  cnt = (float*)(ws + off);  off += (size_t)N_NODE * sizeof(float);
  __bf16* wl  = (__bf16*)(ws + off); off += 128 * 128 * sizeof(__bf16);
  __bf16* wr  = (__bf16*)(ws + off); off += 128 * 128 * sizeof(__bf16);
  __bf16* w1  = (__bf16*)(ws + off); off += 64 * 128 * sizeof(__bf16);
  __bf16* w2  = (__bf16*)(ws + off); off += 32 * 64 * sizeof(__bf16);

  // 1) weights -> bf16
  k_cvt_weights<<<(128 * 128 + 255) / 256, 256, 0, stream>>>(W_l, W_r, W1, W2,
                                                             wl, wr, w1, w2);
  // 2) zero agg + cnt (contiguous)
  long nz = (long)N_NODE * FEAT + N_NODE;
  k_zero<<<(int)((nz + 255) / 256), 256, 0, stream>>>(agg, nz);
  // 3) x = n_fea[src] * n_fea[dst]  (bf16, wave-per-edge)
  k_edge_feat<<<NEDGE / 8, 256, 0, stream>>>(rna, prot, ei, xbf);
  // 4) scatter mean-aggregation inputs
  k_scatter<<<NEDGE / 8, 256, 0, stream>>>(xbf, ei, agg, cnt);
  // 5) fused SAGE + MLP + log_softmax (WMMA)
  k_fused<<<(NEDGE / 16) / 8, 256, 0, stream>>>(xbf, agg, cnt, wl, wr, w1, w2,
                                                b_l, b1, b2, W3, b3, out);
  // 6) labels
  k_labels<<<(NEDGE + 255) / 256, 256, 0, stream>>>(out);
}
